// SwinTransformerBlock_22101901705462
// MI455X (gfx1250) — compile-verified, matrix operand order checked by reading, not run on hardware
//
#include <hip/hip_runtime.h>
#include <math.h>

// ---------------------------------------------------------------------------
// Types for CDNA5 WMMA (wave32): V_WMMA_F32_16X16X32_BF16
// ---------------------------------------------------------------------------
typedef __bf16 bf16_t;
typedef __attribute__((ext_vector_type(16))) bf16_t v16bf;
typedef __attribute__((ext_vector_type(8)))  bf16_t v8bf;
typedef __attribute__((ext_vector_type(8)))  float  v8f;

#define DIM    768
#define HEADS  12
#define HD     64
#define WS     64
#define NTOK   12288
#define NWIN   192            // windows per batch element
#define BATCH  2
#define MTOT   (BATCH * NTOK) // 24576 token rows
#define SHIFT  32
#define SCALE  0.125f         // 64^-0.5

union FragU { v16bf v; v8bf h[2]; };

// A-fragment (16x32 bf16, row-major source with leading dim `ld`):
//  lane = 16*hi + r holds row (row0+r), K = {k0+hi*8 .. +7} U {k0+16+hi*8 .. +7}
//  -> two contiguous 16B chunks per lane (b128 loads).
// B-fragment uses the same pattern with "rows" = output columns (weights stored
// row-major as [out, in], i.e. B loaded column-major as the ISA requires).
__device__ __forceinline__ v16bf load_frag(const bf16_t* base,
                                           int ld, int row0, int k0, int lane) {
    const int r  = lane & 15;
    const int hi = (lane >> 4) & 1;
    const bf16_t* p = base + (size_t)(row0 + r) * (size_t)ld + (size_t)(k0 + hi * 8);
    FragU u;
    u.h[0] = *(const v8bf*)(p);
    u.h[1] = *(const v8bf*)(p + 16);
    return u.v;
}

__device__ __forceinline__ v8f wmma_bf16(v16bf a, v16bf b, v8f c) {
    // (neg_a, A, neg_b, B, c_mod, C, reuse_a, reuse_b)
    return __builtin_amdgcn_wmma_f32_16x16x32_bf16(false, a, false, b,
                                                   (short)0, c, false, false);
}

// NEST (Morton) position -> (i, j) on the 8x8 window grid.
__device__ __forceinline__ void nest2ij(int p, int& i, int& j) {
    i = ((p >> 1) & 1) | ((p >> 2) & 2) | ((p >> 3) & 4);
    j = (p & 1) | ((p >> 1) & 2) | ((p >> 2) & 4);
}

// ---------------------------------------------------------------------------
// Elementwise f32 -> bf16 (weight conversion)
// ---------------------------------------------------------------------------
__global__ __launch_bounds__(256)
void cvt_bf16_kernel(const float* __restrict__ in, bf16_t* __restrict__ out, int n) {
    int i = blockIdx.x * 256 + threadIdx.x;
    if (i < n) out[i] = (bf16_t)in[i];
}

// ---------------------------------------------------------------------------
// LayerNorm (+ optional cyclic roll by `shift`) -> bf16
// One block per output token row.
// ---------------------------------------------------------------------------
__global__ __launch_bounds__(256)
void ln_roll_kernel(const float* __restrict__ src, const float* __restrict__ g,
                    const float* __restrict__ b, bf16_t* __restrict__ dst, int shift) {
    __shared__ float red[256];
    const int tid   = threadIdx.x;
    const int row   = blockIdx.x;
    const int batch = row / NTOK;
    const int t     = row - batch * NTOK;
    const int srow  = batch * NTOK + ((t + shift) % NTOK);
    const float* p  = src + (size_t)srow * DIM;

    float x0 = p[tid], x1 = p[tid + 256], x2 = p[tid + 512];
    red[tid] = x0 + x1 + x2;
    __syncthreads();
    #pragma unroll
    for (int o = 128; o > 0; o >>= 1) {
        if (tid < o) red[tid] += red[tid + o];
        __syncthreads();
    }
    const float mean = red[0] * (1.0f / (float)DIM);
    __syncthreads();
    float d0 = x0 - mean, d1 = x1 - mean, d2 = x2 - mean;
    red[tid] = d0 * d0 + d1 * d1 + d2 * d2;
    __syncthreads();
    #pragma unroll
    for (int o = 128; o > 0; o >>= 1) {
        if (tid < o) red[tid] += red[tid + o];
        __syncthreads();
    }
    const float rstd = rsqrtf(red[0] * (1.0f / (float)DIM) + 1e-5f);
    bf16_t* q = dst + (size_t)row * DIM;
    q[tid]       = (bf16_t)(d0 * rstd * g[tid]       + b[tid]);
    q[tid + 256] = (bf16_t)(d1 * rstd * g[tid + 256] + b[tid + 256]);
    q[tid + 512] = (bf16_t)(d2 * rstd * g[tid + 512] + b[tid + 512]);
}

// ---------------------------------------------------------------------------
// Generic WMMA GEMM:  C[M,N] = A[M,K] @ B[N,K]^T + bias[N], epilogue variants.
// Block = 256 threads = 8 waves; block tile 128(M) x 128(N); K-step 32.
// LDS-staged, double-buffered, manually 2x-unrolled pipeline (K/32 is even for
// all call sites): each half addresses a compile-time-constant LDS buffer.
// global_load_b128 -> ds_store_b128 staging, ds_load_b128 fragments,
// global_prefetch_b8 two k-steps ahead, one barrier per k-step.
// ---------------------------------------------------------------------------
#define EPI_QKV  0   // scatter to Q (row-major), K (row-major), V (transposed), all bf16
#define EPI_PROJ 1   // out = shortcut(x at un-rolled index) + acc + bias  (f32)
#define EPI_GELU 2   // bf16( gelu(acc + bias) )
#define EPI_RES  3   // out[m,n] += acc + bias  (f32, in place)

#define LDSS 40      // LDS row stride (32 + 8 pad): 80B spreads the 64 banks

template <int EPI>
__global__ __launch_bounds__(256)
void gemm_wmma_kernel(const bf16_t* __restrict__ A, const bf16_t* __restrict__ Bw,
                      const float* __restrict__ bias, int K,
                      bf16_t* __restrict__ o0, bf16_t* __restrict__ o1,
                      bf16_t* __restrict__ o2,
                      float* __restrict__ fo, const float* __restrict__ fres) {
    __shared__ alignas(16) bf16_t As[2][128][LDSS];
    __shared__ alignas(16) bf16_t Bs[2][128][LDSS];

    const int tid  = threadIdx.x;
    const int lane = tid & 31;
    const int wave = tid >> 5;
    const int wm   = wave >> 1;          // 0..3  (M)
    const int wn   = wave & 1;           // 0..1  (N)
    const int bm   = blockIdx.x * 128;
    const int bn   = blockIdx.y * 128;

    // staging coordinates: thread copies rows {srow, srow+64}, cols [skc, skc+8)
    const int srow = tid >> 2;           // 0..63
    const int skc  = (tid & 3) * 8;      // 0,8,16,24
    const bf16_t* Ag = A  + (size_t)(bm + srow) * K + skc;
    const bf16_t* Bg = Bw + (size_t)(bn + srow) * K + skc;
    const size_t rstep = (size_t)64 * (size_t)K;

    v8f acc[2][4];
    #pragma unroll
    for (int i = 0; i < 2; ++i)
        #pragma unroll
        for (int j = 0; j < 4; ++j)
            acc[i][j] = (v8f){0.f, 0.f, 0.f, 0.f, 0.f, 0.f, 0.f, 0.f};

    v8bf ra0, ra1, rb0, rb1;
    auto fetch = [&](int kk) {
        ra0 = *(const v8bf*)(Ag + kk);
        ra1 = *(const v8bf*)(Ag + rstep + kk);
        rb0 = *(const v8bf*)(Bg + kk);
        rb1 = *(const v8bf*)(Bg + rstep + kk);
    };
    auto stage = [&](int nb) {
        *(v8bf*)&As[nb][srow][skc]      = ra0;
        *(v8bf*)&As[nb][srow + 64][skc] = ra1;
        *(v8bf*)&Bs[nb][srow][skc]      = rb0;
        *(v8bf*)&Bs[nb][srow + 64][skc] = rb1;
    };
    auto compute = [&](const bf16_t* Asb, const bf16_t* Bsb) {
        v16bf a0 = load_frag(Asb, LDSS, wm * 32,      0, lane);
        v16bf a1 = load_frag(Asb, LDSS, wm * 32 + 16, 0, lane);
        v16bf b0 = load_frag(Bsb, LDSS, wn * 64,      0, lane);
        v16bf b1 = load_frag(Bsb, LDSS, wn * 64 + 16, 0, lane);
        v16bf b2 = load_frag(Bsb, LDSS, wn * 64 + 32, 0, lane);
        v16bf b3 = load_frag(Bsb, LDSS, wn * 64 + 48, 0, lane);
        acc[0][0] = wmma_bf16(a0, b0, acc[0][0]);
        acc[0][1] = wmma_bf16(a0, b1, acc[0][1]);
        acc[0][2] = wmma_bf16(a0, b2, acc[0][2]);
        acc[0][3] = wmma_bf16(a0, b3, acc[0][3]);
        acc[1][0] = wmma_bf16(a1, b0, acc[1][0]);
        acc[1][1] = wmma_bf16(a1, b1, acc[1][1]);
        acc[1][2] = wmma_bf16(a1, b2, acc[1][2]);
        acc[1][3] = wmma_bf16(a1, b3, acc[1][3]);
    };

    // preload k-step 0 into buffer 0
    fetch(0);
    stage(0);
    __syncthreads();

    const int nsteps = K >> 5;           // even (24 or 96) at every call site
    for (int s = 0; s < nsteps; s += 2) {
        // ---- step s: consume buf0, stage buf1 (s+1 < nsteps always holds) --
        fetch((s + 1) << 5);
        if (s + 2 < nsteps) {
            __builtin_prefetch(Ag + ((s + 2) << 5), 0, 1);   // global_prefetch_b8
            __builtin_prefetch(Bg + ((s + 2) << 5), 0, 1);
        }
        compute(&As[0][0][0], &Bs[0][0][0]);
        stage(1);
        __syncthreads();

        // ---- step s+1: consume buf1, stage buf0 ---------------------------
        if (s + 2 < nsteps) {
            fetch((s + 2) << 5);
            if (s + 3 < nsteps) {
                __builtin_prefetch(Ag + ((s + 3) << 5), 0, 1);
                __builtin_prefetch(Bg + ((s + 3) << 5), 0, 1);
            }
        }
        compute(&As[1][0][0], &Bs[1][0][0]);
        if (s + 2 < nsteps) {
            stage(0);
            __syncthreads();
        }
    }

    // C/D layout: vgpr rr, lanes 0-15 -> M = rr, N = lane; lanes 16-31 -> M = 8+rr.
    const int half = lane >> 4;
    const int nl   = lane & 15;
    #pragma unroll
    for (int i = 0; i < 2; ++i) {
        const int m0 = bm + wm * 32 + i * 16;
        #pragma unroll
        for (int j = 0; j < 4; ++j) {
            const int n0 = bn + wn * 64 + j * 16;
            #pragma unroll
            for (int rr = 0; rr < 8; ++rr) {
                const int m = m0 + half * 8 + rr;
                const int n = n0 + nl;
                float val = acc[i][j][rr] + bias[n];
                if constexpr (EPI == EPI_QKV) {
                    const int which = n / DIM;
                    const int rem   = n - which * DIM;
                    const int h     = rem >> 6;
                    const int d     = rem & 63;
                    const int win   = m >> 6;
                    const int t     = m & 63;
                    const size_t base = ((size_t)(win * HEADS + h)) * (WS * HD);
                    const bf16_t bv = (bf16_t)val;
                    if (which == 0)      o0[base + t * HD + d] = bv;     // Q row-major
                    else if (which == 1) o1[base + t * HD + d] = bv;     // K row-major
                    else                 o2[base + d * WS + t] = bv;     // V transposed
                } else if constexpr (EPI == EPI_PROJ) {
                    const int b  = m / NTOK;
                    const int ts = m - b * NTOK;
                    const int t  = (ts + SHIFT) % NTOK;                  // roll back
                    const size_t oi = ((size_t)(b * NTOK + t)) * DIM + n;
                    fo[oi] = fres[oi] + val;                             // shortcut + proj
                } else if constexpr (EPI == EPI_GELU) {
                    const float gl = 0.5f * val * (1.0f + erff(val * 0.70710678118654752f));
                    o0[(size_t)m * (4 * DIM) + n] = (bf16_t)gl;
                } else { // EPI_RES
                    const size_t oi = (size_t)m * DIM + n;
                    fo[oi] += val;
                }
            }
        }
    }
}

// ---------------------------------------------------------------------------
// Windowed attention: one block per (window, head). 4 waves x 16 rows.
// S = (Q K^T) * scale + rel_bias + mask ; softmax ; O = P V.
// Window-head tiles are 8KB and L1-resident -> direct fragment loads.
// ---------------------------------------------------------------------------
__global__ __launch_bounds__(128)
void attn_wmma_kernel(const bf16_t* __restrict__ qb, const bf16_t* __restrict__ kb,
                      const bf16_t* __restrict__ vb, const float* __restrict__ relt,
                      bf16_t* __restrict__ ob) {
    __shared__ alignas(16) bf16_t pbuf[4][16][WS];

    const int win  = blockIdx.x;     // 0 .. BATCH*NWIN-1
    const int head = blockIdx.y;
    const int tid  = threadIdx.x;
    const int lane = tid & 31;
    const int wave = tid >> 5;
    const int mrow = wave * 16;
    const int half = lane >> 4;
    const int nl   = lane & 15;

    const size_t whbase = ((size_t)(win * HEADS + head)) * (WS * HD);
    const bf16_t* qp = qb + whbase;
    const bf16_t* kp = kb + whbase;
    const bf16_t* vp = vb + whbase;

    // ---- S = Q K^T --------------------------------------------------------
    v8f s[4];
    #pragma unroll
    for (int j = 0; j < 4; ++j) s[j] = (v8f){0.f,0.f,0.f,0.f,0.f,0.f,0.f,0.f};
    #pragma unroll
    for (int kk = 0; kk < HD; kk += 32) {
        v16bf aq = load_frag(qp, HD, mrow, kk, lane);
        #pragma unroll
        for (int j = 0; j < 4; ++j) {
            v16bf bk = load_frag(kp, HD, j * 16, kk, lane);  // B col n = K row n
            s[j] = wmma_bf16(aq, bk, s[j]);
        }
    }

    // ---- scale + relative position bias + shifted-window mask -------------
    const bool maskwin = ((win % NWIN) == (NWIN - 1)); // only last window mixes wrap
    #pragma unroll
    for (int rr = 0; rr < 8; ++rr) {
        const int m = mrow + half * 8 + rr;   // query pos in window (NEST order)
        int im, jm; nest2ij(m, im, jm);
        #pragma unroll
        for (int j = 0; j < 4; ++j) {
            const int n = j * 16 + nl;        // key pos in window (NEST order)
            int in_, jn; nest2ij(n, in_, jn);
            const int idx = (im - in_ + 7) * 15 + (jm - jn + 7);
            float v = s[j][rr] * SCALE + relt[idx * HEADS + head];
            if (maskwin && ((m < SHIFT) != (n < SHIFT))) v -= 100.0f;
            s[j][rr] = v;
        }
    }

    // ---- softmax over the 64 keys of each row ------------------------------
    // Row m lives in 16 lanes (fixed half) x 4 accumulators at vgpr rr.
    // xor-shuffles by 1/2/4/8 stay inside the 16-lane half-group.
    #pragma unroll
    for (int rr = 0; rr < 8; ++rr) {
        float mx = fmaxf(fmaxf(s[0][rr], s[1][rr]), fmaxf(s[2][rr], s[3][rr]));
        #pragma unroll
        for (int off = 8; off >= 1; off >>= 1) mx = fmaxf(mx, __shfl_xor(mx, off, 32));
        float sum = 0.f;
        #pragma unroll
        for (int j = 0; j < 4; ++j) {
            float e = __expf(s[j][rr] - mx);
            s[j][rr] = e;
            sum += e;
        }
        #pragma unroll
        for (int off = 8; off >= 1; off >>= 1) sum += __shfl_xor(sum, off, 32);
        const float inv = 1.0f / sum;
        #pragma unroll
        for (int j = 0; j < 4; ++j) s[j][rr] *= inv;
    }

    // ---- stage P (bf16) to LDS so it can be reloaded as an A fragment ------
    #pragma unroll
    for (int rr = 0; rr < 8; ++rr)
        #pragma unroll
        for (int j = 0; j < 4; ++j)
            pbuf[wave][half * 8 + rr][j * 16 + nl] = (bf16_t)s[j][rr];
    __syncthreads();

    // ---- O = P V  (B col d = V^T row d = vb row d) -------------------------
    v8f o[4];
    #pragma unroll
    for (int j = 0; j < 4; ++j) o[j] = (v8f){0.f,0.f,0.f,0.f,0.f,0.f,0.f,0.f};
    #pragma unroll
    for (int kk = 0; kk < WS; kk += 32) {
        v16bf ap = load_frag(&pbuf[wave][0][0], WS, 0, kk, lane);
        #pragma unroll
        for (int j = 0; j < 4; ++j) {
            v16bf bv = load_frag(vp, WS, j * 16, kk, lane);
            o[j] = wmma_bf16(ap, bv, o[j]);
        }
    }

    // ---- write context (bf16) in [win][t][DIM] layout for the proj GEMM ----
    #pragma unroll
    for (int rr = 0; rr < 8; ++rr) {
        const int t = mrow + half * 8 + rr;
        #pragma unroll
        for (int j = 0; j < 4; ++j) {
            const int d = j * 16 + nl;
            ob[((size_t)win * WS + t) * DIM + head * HD + d] = (bf16_t)o[j][rr];
        }
    }
}

// ---------------------------------------------------------------------------
// Host-side orchestration
// ---------------------------------------------------------------------------
extern "C" void kernel_launch(void* const* d_in, const int* in_sizes, int n_in,
                              void* d_out, int out_size, void* d_ws, size_t ws_size,
                              hipStream_t stream) {
    (void)in_sizes; (void)n_in; (void)out_size; (void)ws_size;

    const float* x      = (const float*)d_in[0];
    const float* n1g    = (const float*)d_in[1];
    const float* n1b    = (const float*)d_in[2];
    const float* qkv_w  = (const float*)d_in[3];
    const float* qkv_b  = (const float*)d_in[4];
    const float* relt   = (const float*)d_in[5];
    const float* proj_w = (const float*)d_in[6];
    const float* proj_b = (const float*)d_in[7];
    const float* n2g    = (const float*)d_in[8];
    const float* n2b    = (const float*)d_in[9];
    const float* fc1_w  = (const float*)d_in[10];
    const float* fc1_b  = (const float*)d_in[11];
    const float* fc2_w  = (const float*)d_in[12];
    const float* fc2_b  = (const float*)d_in[13];
    float* out = (float*)d_out;

    // Workspace layout (bytes), all 256B aligned:
    //   hbf  : MTOT*DIM bf16            (normalized activations; reused for LN2)
    //   qb/kb/vb/ob : 4 * MTOT*DIM bf16 (attention region; aliased by fc1 act m1)
    //   wq/wp/w1/w2 : bf16 weights
    char* ws = (char*)d_ws;
    const size_t actB = (size_t)MTOT * DIM * sizeof(bf16_t);   // 37,748,736 B
    bf16_t* hbf = (bf16_t*)ws;
    bf16_t* qb  = (bf16_t*)(ws + actB);
    bf16_t* kb  = qb + (size_t)MTOT * DIM;
    bf16_t* vb  = kb + (size_t)MTOT * DIM;
    bf16_t* ob  = vb + (size_t)MTOT * DIM;
    bf16_t* m1  = qb;  // fc1 activations (24576x3072 bf16) alias q/k/v/o exactly
    bf16_t* wq  = (bf16_t*)(ws + actB + 4 * actB);
    bf16_t* wp  = wq + (size_t)3 * DIM * DIM;
    bf16_t* w1  = wp + (size_t)DIM * DIM;
    bf16_t* w2  = w1 + (size_t)4 * DIM * DIM;

    // 1) weight conversion f32 -> bf16
    {
        int n;
        n = 3 * DIM * DIM; cvt_bf16_kernel<<<(n + 255) / 256, 256, 0, stream>>>(qkv_w, wq, n);
        n = DIM * DIM;     cvt_bf16_kernel<<<(n + 255) / 256, 256, 0, stream>>>(proj_w, wp, n);
        n = 4 * DIM * DIM; cvt_bf16_kernel<<<(n + 255) / 256, 256, 0, stream>>>(fc1_w, w1, n);
        n = 4 * DIM * DIM; cvt_bf16_kernel<<<(n + 255) / 256, 256, 0, stream>>>(fc2_w, w2, n);
    }

    // 2) LN1 + cyclic shift (-SHIFT) -> hbf (bf16)
    ln_roll_kernel<<<MTOT, 256, 0, stream>>>(x, n1g, n1b, hbf, SHIFT);

    // 3) QKV GEMM: [24576,768] @ [2304,768]^T, scatter into Q/K/V^T tiles
    gemm_wmma_kernel<EPI_QKV><<<dim3(MTOT / 128, (3 * DIM) / 128), 256, 0, stream>>>(
        hbf, wq, qkv_b, DIM, qb, kb, vb, nullptr, nullptr);

    // 4) windowed attention (WMMA QK^T + softmax + WMMA PV)
    attn_wmma_kernel<<<dim3(BATCH * NWIN, HEADS), 128, 0, stream>>>(qb, kb, vb, relt, ob);

    // 5) proj GEMM + un-shift + shortcut -> d_out (f32)
    gemm_wmma_kernel<EPI_PROJ><<<dim3(MTOT / 128, DIM / 128), 256, 0, stream>>>(
        ob, wp, proj_b, DIM, nullptr, nullptr, nullptr, out, x);

    // 6) LN2 (no shift) on d_out -> hbf (bf16)
    ln_roll_kernel<<<MTOT, 256, 0, stream>>>(out, n2g, n2b, hbf, 0);

    // 7) FC1 GEMM + exact GELU -> m1 (bf16), aliasing the attention region
    gemm_wmma_kernel<EPI_GELU><<<dim3(MTOT / 128, (4 * DIM) / 128), 256, 0, stream>>>(
        hbf, w1, fc1_b, DIM, m1, nullptr, nullptr, nullptr, nullptr);

    // 8) FC2 GEMM + residual add in-place on d_out (f32)
    gemm_wmma_kernel<EPI_RES><<<dim3(MTOT / 128, DIM / 128), 256, 0, stream>>>(
        m1, w2, fc2_b, 4 * DIM, nullptr, nullptr, nullptr, out, nullptr);
}
